// CoreAttention_31327491457426
// MI455X (gfx1250) — matrix operand order repaired and checked by reading, not compiled, not run on hardware
//
#include <hip/hip_runtime.h>
#include <hip/hip_bf16.h>

typedef __attribute__((ext_vector_type(16))) _Float16 v16h;
typedef __attribute__((ext_vector_type(8)))  float    v8f;

#define HEADS 8
#define HDIM  32
#define ROW   256                    // HEADS*HDIM
#define INV_SCALE 0.1767766952966369f // 1/sqrt(32)

__global__ void k_zero_i32(int* __restrict__ p, int n) {
  int i = blockIdx.x * blockDim.x + threadIdx.x;
  if (i < n) p[i] = 0;
}

__global__ void k_hist(const int* __restrict__ eidx, int* __restrict__ deg, int E) {
  int e = blockIdx.x * blockDim.x + threadIdx.x;
  if (e < E) atomicAdd(&deg[eidx[E + e]], 1);
}

// Single-workgroup exclusive scan of deg[0..n) -> off[0..n], also copies to cur.
__global__ void k_scan(const int* __restrict__ deg, int* __restrict__ off,
                       int* __restrict__ cur, int n) {
  __shared__ int sd[1024];
  __shared__ int carry;
  const int tid = threadIdx.x;
  if (tid == 0) carry = 0;
  __syncthreads();
  for (int base = 0; base < n; base += 1024) {
    int i   = base + tid;
    int val = (i < n) ? deg[i] : 0;
    sd[tid] = val;
    __syncthreads();
    for (int o = 1; o < 1024; o <<= 1) {
      int t = (tid >= o) ? sd[tid - o] : 0;
      __syncthreads();
      sd[tid] += t;
      __syncthreads();
    }
    int excl = sd[tid] - val + carry;
    if (i < n) { off[i] = excl; cur[i] = excl; }
    __syncthreads();
    if (tid == 0) carry += sd[1023];
    __syncthreads();
  }
  if (tid == 0) off[n] = carry;
}

__global__ void k_scatter(const int* __restrict__ eidx, int* __restrict__ cur,
                          int* __restrict__ ssrc, int E) {
  int e = blockIdx.x * blockDim.x + threadIdx.x;
  if (e < E) {
    int dst = eidx[E + e];
    int pos = atomicAdd(&cur[dst], 1);
    ssrc[pos] = eidx[e];   // src node of this edge, grouped by dst
  }
}

// One wave32 per destination node. Edges processed in groups of 16.
// Per head: V_WMMA_F32_16X16X32_F16 with A = 16 gathered K rows (16x32),
// B = q[dst] head h broadcast into all 16 columns -> every column of D is
// the per-edge score vector. Softmax-weighted V accumulation in registers.
__global__ void __launch_bounds__(256)
k_attn(const float* __restrict__ q, const float* __restrict__ k,
       const float* __restrict__ v, const int* __restrict__ off,
       const int* __restrict__ ssrc, float* __restrict__ out, int n) {
  const int wv   = (blockIdx.x * blockDim.x + threadIdx.x) >> 5;
  const int lane = threadIdx.x & 31;
  if (wv >= n) return;                       // wave-uniform branch
  const int dst = wv;
  const int beg = __builtin_amdgcn_readfirstlane(off[dst]);
  const int end = __builtin_amdgcn_readfirstlane(off[dst + 1]);

  const int qb    = (lane < 16) ? 0 : 16;    // B-matrix K-block base
  const int koff  = (lane < 16) ? 0 : 8;     // unused; A also indexes k from qb
  const int am    = lane & 15;               // A-matrix row owned by lane
  const int rbase = (lane < 16) ? 0 : 8;     // D rows held by this lane

  float acc[HEADS];
  float zed[HEADS];
  #pragma unroll
  for (int h = 0; h < HEADS; ++h) { acc[h] = 0.f; zed[h] = 0.f; }

  const float* qrow = q + (long)dst * ROW;

  for (int g = beg; g < end; g += 16) {
    const int cnt = (end - g < 16) ? (end - g) : 16;
    int sidx[16];
    #pragma unroll
    for (int m = 0; m < 16; ++m) {
      int mm = (m < cnt) ? m : (cnt - 1);    // clamp: padded rows masked later
      sidx[m] = ssrc[g + mm];
    }
    const float* krow = k + (long)sidx[am] * ROW;

    #pragma unroll
    for (int h = 0; h < HEADS; ++h) {
      // A: gathered K tile, 16 edges x 32 dims, f16
      v16h a;
      #pragma unroll
      for (int i = 0; i < 8; ++i) {
        a[i]     = (_Float16)krow[h * HDIM + qb + i];
        a[8 + i] = (_Float16)krow[h * HDIM + qb + 8 + i];
      }
      // B: q[dst] head h broadcast into all 16 columns
      v16h b;
      #pragma unroll
      for (int i = 0; i < 16; ++i)
        b[i] = (_Float16)qrow[h * HDIM + qb + i];

      v8f c = {};
      c = __builtin_amdgcn_wmma_f32_16x16x32_f16(false, a, false, b,
                                                 (short)0, c, false, false);

      // Every column identical -> lane holds scores for rows rbase..rbase+7
      float mine[8];
      #pragma unroll
      for (int j = 0; j < 8; ++j) {
        float s = c[j] * INV_SCALE;
        s = fminf(5.f, fmaxf(-5.f, s));
        float e = __expf(s);
        if (rbase + j >= cnt) e = 0.f;       // mask padded edges
        mine[j] = e;
      }
      // Exchange halves (SWAPX16) so every lane has all 16 edge weights
      float w[16];
      #pragma unroll
      for (int j = 0; j < 8; ++j) {
        float oth = __int_as_float(
            __builtin_amdgcn_ds_swizzle(__float_as_int(mine[j]), 0x401F));
        w[j]     = (lane < 16) ? mine[j] : oth;
        w[8 + j] = (lane < 16) ? oth     : mine[j];
      }
      // Weighted V accumulation: lane owns dim d = lane of head h
      float zl = 0.f, al = 0.f;
      #pragma unroll
      for (int m = 0; m < 16; ++m) {
        zl += w[m];
        al  = fmaf(w[m], v[(long)sidx[m] * ROW + h * HDIM + lane], al);
      }
      zed[h] += zl;
      acc[h] += al;
    }
  }

  float* orow = out + (long)dst * ROW;
  #pragma unroll
  for (int h = 0; h < HEADS; ++h)
    orow[h * HDIM + lane] = acc[h] / (zed[h] + 1e-6f);
}

extern "C" void kernel_launch(void* const* d_in, const int* in_sizes, int n_in,
                              void* d_out, int out_size, void* d_ws, size_t ws_size,
                              hipStream_t stream) {
  const float* q    = (const float*)d_in[0];
  const float* k    = (const float*)d_in[1];
  const float* v    = (const float*)d_in[2];
  const int*   eidx = (const int*)d_in[3];   // [2,E] flat: [0..E)=src, [E..2E)=dst

  const int N = in_sizes[0] / ROW;           // 20000
  const int E = in_sizes[3] / 2;             // 320000

  int* deg  = (int*)d_ws;                    // N
  int* off  = deg + N;                       // N+1
  int* cur  = off + (N + 1);                 // N
  int* ssrc = cur + N;                       // E
  float* out = (float*)d_out;

  k_zero_i32<<<(N + 255) / 256, 256, 0, stream>>>(deg, N);
  k_hist    <<<(E + 255) / 256, 256, 0, stream>>>(eidx, deg, E);
  k_scan    <<<1, 1024, 0, stream>>>(deg, off, cur, N);
  k_scatter <<<(E + 255) / 256, 256, 0, stream>>>(eidx, cur, ssrc, E);
  k_attn    <<<(N + 7) / 8, 256, 0, stream>>>(q, k, v, off, ssrc, out, N);
}
